// BasisSlotAffinityGAT_26749056320340
// MI455X (gfx1250) — matrix operand order repaired and checked by reading, not compiled, not run on hardware
//
#include <hip/hip_runtime.h>
#include <hip/hip_bf16.h>

// ---------------------------------------------------------------------------
// BasisSlotAffinityGAT for MI455X (gfx1250, wave32, WMMA).
// B=32, N=256, D=256, H=8, K=16.
// Bandwidth-bound: 134 MB of output vs 2.8 GFLOP. Strategy: fuse everything,
// write Q / bias_log exactly once, use WMMA for the two GEMM stages.
// ---------------------------------------------------------------------------

typedef _Float16 v16h __attribute__((ext_vector_type(16)));
typedef float    v8f  __attribute__((ext_vector_type(8)));
typedef float    v2f  __attribute__((ext_vector_type(2)));

#define Bq   32
#define Nq   256
#define Dq   256
#define Hq   8
#define Kq   16
#define BHNN 16777216   // 32*8*256*256

// ===========================================================================
// K1: z = [desc|nv] @ Wf^T + bf ; S = softmax(z @ Ws^T + bs)
// One block = 16 rows of z. 8 waves; wave w owns z columns [32w,32w+32).
// WMMA f32_16x16x32_f16: K-loop over 512 in steps of 32.
// ===========================================================================
__global__ __launch_bounds__(256)
void k1_fuse_slot(const float* __restrict__ desc, const float* __restrict__ nv,
                  const float* __restrict__ Wf,   const float* __restrict__ bf,
                  const float* __restrict__ Ws,   const float* __restrict__ bs,
                  float* __restrict__ Sout)
{
    const int t    = threadIdx.x;
    const int lane = t & 31;
    const int wave = t >> 5;
    const int rowbase = blockIdx.x * 16;

    __shared__ float z[16][260];     // 16 x 256 z-tile (+pad)
    __shared__ float lgt[16][16];

    // row handled by this lane (A-operand: M = lane%16)
    const int mrow = rowbase + (lane & 15);
    const float* xr_d = desc + (size_t)mrow * Dq;   // columns 0..255
    const float* xr_n = nv   + (size_t)mrow * Dq;   // columns 256..511

    const int n0 = wave * 32;
    const int d0 = n0 + (lane & 15);       // z column for B fragment (tile 0)
    const int d1 = d0 + 16;                // tile 1
    const int hiA = (lane >= 16) ? 8  : 0; // A layout: upper lanes get K+8
    const int hiB = (lane >= 16) ? 8  : 0; // B layout: upper lanes get K+8

    v8f acc0 = {0,0,0,0,0,0,0,0};
    v8f acc1 = {0,0,0,0,0,0,0,0};

    for (int kk = 0; kk < 2 * Dq; kk += 32) {
        const float* xr  = (kk < Dq) ? xr_d : xr_n;
        const int   kloc = kk & (Dq - 1);
        v16h a, b0, b1;
#pragma unroll
        for (int p = 0; p < 8; ++p) {
            // A 16x32 f16 layout: VGPR p holds K = (p/4)*16 + hiA + (p%4)*2 (+1)
            const int kof = ((p >> 2) << 4) + hiA + ((p & 3) << 1);
            v2f xa = *(const v2f*)(xr + kloc + kof);
            a[2*p]   = (_Float16)xa[0];
            a[2*p+1] = (_Float16)xa[1];
            // B 32x16 f16 layout: VGPR p holds K = (p/4)*16 + hiB + (p%4)*2 (+1), col = lane%16
            const int kb = kk + ((p >> 2) << 4) + hiB + ((p & 3) << 1);
            v2f w0 = *(const v2f*)(Wf + (size_t)d0 * (2*Dq) + kb);
            b0[2*p]   = (_Float16)w0[0];
            b0[2*p+1] = (_Float16)w0[1];
            v2f w1 = *(const v2f*)(Wf + (size_t)d1 * (2*Dq) + kb);
            b1[2*p]   = (_Float16)w1[0];
            b1[2*p+1] = (_Float16)w1[1];
        }
        acc0 = __builtin_amdgcn_wmma_f32_16x16x32_f16(false, a, false, b0,
                                                      (short)0, acc0, false, false);
        acc1 = __builtin_amdgcn_wmma_f32_16x16x32_f16(false, a, false, b1,
                                                      (short)0, acc1, false, false);
    }

    // C layout: lanes 0-15 -> rows v, lanes 16-31 -> rows v+8; col = lane%16
    const int rloc = (lane >= 16) ? 8 : 0;
    const int c0   = n0 + (lane & 15);
    const float bia0 = bf[c0];
    const float bia1 = bf[c0 + 16];
#pragma unroll
    for (int v = 0; v < 8; ++v) {
        z[rloc + v][c0]      = acc0[v] + bia0;
        z[rloc + v][c0 + 16] = acc1[v] + bia1;
    }
    __syncthreads();

    // slot logits: thread t -> (row r, slot ks)
    {
        const int r = t >> 4, ks = t & 15;
        const float* wr = Ws + (size_t)ks * Dq;
        float lg = bs[ks];
        for (int d = 0; d < Dq; ++d) lg += z[r][d] * wr[d];
        lgt[r][ks] = lg;
    }
    __syncthreads();
    {
        const int r = t >> 4, ks = t & 15;
        float mx = lgt[r][0];
#pragma unroll
        for (int j = 1; j < Kq; ++j) mx = fmaxf(mx, lgt[r][j]);
        float s = 0.f;
#pragma unroll
        for (int j = 0; j < Kq; ++j) s += __expf(lgt[r][j] - mx);
        Sout[(size_t)(rowbase + r) * Kq + ks] = __expf(lgt[r][ks] - mx) / s;
    }
}

// ===========================================================================
// K2: G = zero_diag(sinkhorn(softmax(G_param))); g_reg; zero accumulators.
// Single block, 256 threads. Whole G lives in LDS (8 KB).
// ===========================================================================
__global__ __launch_bounds__(256)
void k2_gproc(const float* __restrict__ Gp, float* __restrict__ Gout,
              float* __restrict__ scal)
{
    __shared__ float g[Hq][Kq][Kq];
    __shared__ float nrm[Hq], sq[Hq];
    __shared__ float red[256];
    const int t = threadIdx.x;

    for (int i = t; i < Hq*Kq*Kq; i += 256) ((float*)g)[i] = Gp[i];
    __syncthreads();

    // row softmax (G_TEMP = 1), then clamp to SINK_EPS
    if (t < Hq*Kq) {
        float* row = g[t >> 4][t & 15];
        float mx = row[0];
#pragma unroll
        for (int j = 1; j < Kq; ++j) mx = fmaxf(mx, row[j]);
        float s = 0.f;
#pragma unroll
        for (int j = 0; j < Kq; ++j) { float e = __expf(row[j] - mx); row[j] = e; s += e; }
#pragma unroll
        for (int j = 0; j < Kq; ++j) row[j] = fmaxf(row[j] / s, 1e-6f);
    }
    __syncthreads();

    for (int it = 0; it < 10; ++it) {
        if (t < Hq*Kq) {                       // row normalize
            float* row = g[t >> 4][t & 15];
            float s = 1e-6f;
#pragma unroll
            for (int j = 0; j < Kq; ++j) s += row[j];
            float inv = 1.f / s;
#pragma unroll
            for (int j = 0; j < Kq; ++j) row[j] *= inv;
        }
        __syncthreads();
        if (t < Hq*Kq) {                       // column normalize
            const int h = t >> 4, c = t & 15;
            float s = 1e-6f;
#pragma unroll
            for (int i = 0; i < Kq; ++i) s += g[h][i][c];
            float inv = 1.f / s;
#pragma unroll
            for (int i = 0; i < Kq; ++i) g[h][i][c] *= inv;
        }
        __syncthreads();
    }
    if (t < Hq*Kq) g[t >> 4][t & 15][t & 15] = 0.f;   // zero diagonal
    __syncthreads();

    for (int i = t; i < Hq*Kq*Kq; i += 256) Gout[i] = ((float*)g)[i];

    if (t < Hq) {                               // per-head norms
        const float* gh = (const float*)g[t];
        float s = 0.f;
        for (int i = 0; i < Kq*Kq; ++i) s += gh[i] * gh[i];
        sq[t]  = s;
        nrm[t] = fmaxf(sqrtf(s), 1e-8f);
    }
    __syncthreads();
    {                                           // sum(Gram) = |sum_h v_h|^2
        float w = 0.f;
#pragma unroll
        for (int h = 0; h < Hq; ++h) w += ((const float*)g[h])[t] / nrm[h];
        red[t] = w * w;
    }
    __syncthreads();
    for (int off = 128; off; off >>= 1) {
        if (t < off) red[t] += red[t + off];
        __syncthreads();
    }
    if (t == 0) {
        float trace = 0.f;
        for (int h = 0; h < Hq; ++h) trace += sq[h] / (nrm[h] * nrm[h]);
        scal[0] = 0.02f * (red[0] - trace) / (float)(Hq * (Hq - 1));  // g_reg
        scal[1] = 0.f;   // orth accumulator
        scal[2] = 0.f;   // usage-kl accumulator
    }
}

// ===========================================================================
// K3: per (b,h): A = (S G) S^T ; Q = softmax(2A) ; bias_log = log(max(Q,eps)).
// WMMA f32_16x16x4 (K=16 -> 4 steps). 8 waves, wave handles 2 row-strips.
// ===========================================================================
__global__ __launch_bounds__(256)
void k3_affinity(const float* __restrict__ Sws, const float* __restrict__ Gws,
                 float* __restrict__ out)
{
    const int bh = blockIdx.x;
    const int b  = bh >> 3, h = bh & 7;
    const int t  = threadIdx.x, lane = t & 31, wave = t >> 5;

    __shared__ float s[Nq][Kq + 1];
    __shared__ float g[Kq][Kq];

    for (int i = t; i < Nq * Kq; i += 256)
        s[i >> 4][i & 15] = Sws[(size_t)b * Nq * Kq + i];
    g[t >> 4][t & 15] = Gws[h * Kq * Kq + t];
    __syncthreads();

    const size_t base = (size_t)bh * Nq * Nq;
    const int M   = lane & 15;
    const int hi2 = (lane >= 16) ? 2 : 0;   // f32 A/B layout: upper lanes K+2
    const int rofs = (lane >= 16) ? 8 : 0;  // C layout row offset

    for (int sidx = wave; sidx < 16; sidx += 8) {
        const int rowbase = sidx * 16;

        // A fragments of SG (built directly in f32 16x4 A layout)
        v2f afr[4];
#pragma unroll
        for (int s4 = 0; s4 < 4; ++s4) {
            const int l0 = 4 * s4 + hi2;
            float a0 = 0.f, a1 = 0.f;
#pragma unroll
            for (int l = 0; l < Kq; ++l) {
                const float sv = s[rowbase + M][l];
                a0 += sv * g[l][l0];
                a1 += sv * g[l][l0 + 1];
            }
            v2f av = {a0, a1};
            afr[s4] = av;
        }

        // A-strip: 16 col-tiles of SG @ S^T  (B[k=l][n=m] = S[m][l])
        v8f acc[16];
#pragma unroll
        for (int ct = 0; ct < 16; ++ct) {
            v8f c = {0,0,0,0,0,0,0,0};
#pragma unroll
            for (int s4 = 0; s4 < 4; ++s4) {
                const int l0 = 4 * s4 + hi2;
                const int m  = ct * 16 + M;
                v2f bv = { s[m][l0], s[m][l0 + 1] };
                c = __builtin_amdgcn_wmma_f32_16x16x4_f32(false, afr[s4], false, bv,
                                                          (short)0, c, false, false);
            }
            acc[ct] = c;
        }

        // fused row softmax (scale 1/tau = 2) + Q + bias_log
#pragma unroll
        for (int v = 0; v < 8; ++v) {
            float mx = -3.4e38f;
#pragma unroll
            for (int ct = 0; ct < 16; ++ct) mx = fmaxf(mx, acc[ct][v]);
            mx *= 2.0f;
#pragma unroll
            for (int d = 8; d; d >>= 1) mx = fmaxf(mx, __shfl_xor(mx, d, 16));
            float se = 0.f;
#pragma unroll
            for (int ct = 0; ct < 16; ++ct) se += __expf(2.0f * acc[ct][v] - mx);
#pragma unroll
            for (int d = 8; d; d >>= 1) se += __shfl_xor(se, d, 16);
            const float inv = 1.0f / se;
            const int row = rowbase + rofs + v;
            const size_t ro = base + (size_t)row * Nq;
#pragma unroll
            for (int ct = 0; ct < 16; ++ct) {
                const float q = __expf(2.0f * acc[ct][v] - mx) * inv;
                const int col = ct * 16 + M;
                out[BHNN + ro + col] = q;                       // Q (2nd output)
                out[ro + col] = __logf(fmaxf(q, 1e-8f));       // bias_log (1st)
            }
        }
    }
}

// ===========================================================================
// K4: per-batch reg_orth + usage-KL partials -> atomic accumulators.
// ===========================================================================
__global__ __launch_bounds__(256)
void k4_reg(const float* __restrict__ Sws, float* __restrict__ scal)
{
    const int b = blockIdx.x, t = threadIdx.x;
    __shared__ float s[Nq][Kq];
    __shared__ float red[256];
    __shared__ float u[Kq];

    for (int i = t; i < Nq * Kq; i += 256) ((float*)s)[i] = Sws[(size_t)b * Nq * Kq + i];
    __syncthreads();

    // StS off-diagonal squared
    {
        const int k = t >> 4, l = t & 15;
        float acc = 0.f;
        for (int n = 0; n < Nq; ++n) acc += s[n][k] * s[n][l];
        acc *= (1.0f / (float)Nq);
        red[t] = (k == l) ? 0.f : acc * acc;
    }
    __syncthreads();
    for (int off = 128; off; off >>= 1) {
        if (t < off) red[t] += red[t + off];
        __syncthreads();
    }
    if (t == 0) atomicAdd(&scal[1], red[0]);

    // slot usage
    if (t < Kq) {
        float su = 0.f;
        for (int n = 0; n < Nq; ++n) su += s[n][t];
        u[t] = su / (float)Nq;
    }
    __syncthreads();
    if (t == 0) {
        float su = 1e-8f;
#pragma unroll
        for (int k = 0; k < Kq; ++k) su += u[k];
        float kl = 0.f;
#pragma unroll
        for (int k = 0; k < Kq; ++k) {
            const float uc = fmaxf(u[k] / su, 1e-8f);
            kl += uc * (__logf(uc) + 2.7725887222397812f);   // - log(1/16)
        }
        atomicAdd(&scal[2], kl);
    }
}

__global__ void k5_final(const float* __restrict__ scal, float* __restrict__ out_total)
{
    if (threadIdx.x == 0 && blockIdx.x == 0)
        *out_total = scal[0]
                   + 0.1f * scal[1] / (float)(Bq * Kq * Kq)
                   + 0.1f * scal[2] / (float)Bq;
}

// ===========================================================================
extern "C" void kernel_launch(void* const* d_in, const int* in_sizes, int n_in,
                              void* d_out, int out_size, void* d_ws, size_t ws_size,
                              hipStream_t stream)
{
    const float* desc = (const float*)d_in[0];
    const float* nv   = (const float*)d_in[1];
    const float* Wf   = (const float*)d_in[2];
    const float* bf   = (const float*)d_in[3];
    const float* Wsl  = (const float*)d_in[4];
    const float* bsl  = (const float*)d_in[5];
    const float* Gp   = (const float*)d_in[6];
    float* out = (float*)d_out;

    float* ws   = (float*)d_ws;
    float* Sws  = ws;                    // B*N*K = 131072 floats
    float* Gws  = ws + 131072;           // H*K*K = 2048 floats
    float* scal = ws + 133120;           // [g_reg, orth_acc, kl_acc]

    k1_fuse_slot<<<(Bq * Nq) / 16, 256, 0, stream>>>(desc, nv, Wf, bf, Wsl, bsl, Sws);
    k2_gproc   <<<1, 256, 0, stream>>>(Gp, Gws, scal);
    k3_affinity<<<Bq * Hq, 256, 0, stream>>>(Sws, Gws, out);
    k4_reg     <<<Bq, 256, 0, stream>>>(Sws, scal);
    k5_final   <<<1, 64, 0, stream>>>(scal, out + (size_t)2 * BHNN);
}
